// CrossAttention_24257975288149
// MI455X (gfx1250) — compile-verified, matrix operand order checked by reading, not run on hardware
//
#include <hip/hip_runtime.h>

// ---------------------------------------------------------------------------
// CrossAttention on gfx1250 (MI455X): bf16 WMMA GEMMs + flash attention.
// ---------------------------------------------------------------------------

typedef __attribute__((ext_vector_type(8)))  __bf16 bf16x8;
typedef __attribute__((ext_vector_type(16))) __bf16 bf16x16;
typedef __attribute__((ext_vector_type(8)))  float  f32x8;

#define DIMM     1024
#define HEADS    16
#define DHEAD    64
#define BATCH    4
#define SEQ      2048
#define ROWS     (BATCH * SEQ)      // 8192
#define KV_LEN   2049               // null token + 2048 context tokens
#define KV_PAD   2080               // round up to 32
#define SCALE_Q  0.125f             // 64^-0.5
#define LN_EPS   1e-5f

__device__ __forceinline__ f32x8 wmma_bf16(bf16x16 a, bf16x16 b, f32x8 c) {
    return __builtin_amdgcn_wmma_f32_16x16x32_bf16(false, a, false, b,
                                                   (short)0, c, false, false);
}

// Load a 16x32 bf16 fragment in the CDNA5 16-bit A/B-transposed layout from a
// row-major matrix (row = lane&15, two 16B runs: K = ksel*8+{0..7} and
// 16+ksel*8+{0..7}, ksel = lane>>4).
__device__ __forceinline__ bf16x16 load_frag(const __bf16* base, int ld, int lane) {
    const int r    = lane & 15;
    const int ksel = (lane >> 4) & 1;
    const __bf16* p = base + (size_t)r * ld + ksel * 8;
    bf16x8 lo = *(const bf16x8*)(p);
    bf16x8 hi = *(const bf16x8*)(p + 16);
    bf16x16 f;
#pragma unroll
    for (int i = 0; i < 8; ++i) { f[i] = lo[i]; f[i + 8] = hi[i]; }
    return f;
}

__device__ __forceinline__ void store_c(float*  p, float v) { *p = v; }
__device__ __forceinline__ void store_c(__bf16* p, float v) { *p = (__bf16)v; }

// ---- 16-lane max reduction via DPP (VALU only, no LDS/ds_bpermute) --------
// dpp_ctrl must be a frontend constant -> template parameter.
template <int CTRL>
__device__ __forceinline__ float maxdpp(float x) {
    int y = __builtin_amdgcn_update_dpp(0, __float_as_int(x), CTRL, 0xF, 0xF, true);
    return fmaxf(x, __int_as_float(y));
}
__device__ __forceinline__ float row16_max(float x) {
    x = maxdpp<0xB1>(x);   // quad_perm(1,0,3,2)  : xor 1
    x = maxdpp<0x4E>(x);   // quad_perm(2,3,0,1)  : xor 2
    x = maxdpp<0x141>(x);  // row_half_mirror     : combine groups of 4
    x = maxdpp<0x140>(x);  // row_mirror          : combine groups of 8
    return x;
}

// ---------------------------------------------------------------------------
// 1) weight transpose + fp32->bf16 convert (optionally scaled): Wt[n,k]=W[k,n]*s
// ---------------------------------------------------------------------------
__global__ void wcvt_t_kernel(const float* __restrict__ W, __bf16* __restrict__ Wt,
                              int K, int N, float scale) {
    size_t i = (size_t)blockIdx.x * blockDim.x + threadIdx.x;
    if (i >= (size_t)K * N) return;
    int k = (int)(i / N), n = (int)(i % N);
    Wt[(size_t)n * K + k] = (__bf16)(W[i] * scale);
}

// plain fp32 -> bf16 convert
__global__ void cvt_bf16_kernel(const float* __restrict__ in, __bf16* __restrict__ out,
                                size_t n) {
    size_t i = (size_t)blockIdx.x * blockDim.x + threadIdx.x;
    if (i < n) out[i] = (__bf16)in[i];
}

// ---------------------------------------------------------------------------
// 2) stable LayerNorm: x /= max(x); mean/var; (x-mean)*rsqrt(var+eps)*g
//    one block per row (1024 cols, 256 threads, 8 waves)
// ---------------------------------------------------------------------------
template <typename OutT>
__global__ __launch_bounds__(256) void ln_stable_kernel(const float* __restrict__ X,
                                                        const float* __restrict__ g,
                                                        OutT* __restrict__ out) {
    const int row  = blockIdx.x;
    const int tid  = threadIdx.x;
    const int lane = tid & 31, wv = tid >> 5;
    const float* xr = X + (size_t)row * DIMM;
    OutT* orow = out + (size_t)row * DIMM;
    __shared__ float redA[8];
    __shared__ float redB[8];

    float v[4];
#pragma unroll
    for (int i = 0; i < 4; ++i) v[i] = xr[tid + i * 256];

    // block max
    float mx = fmaxf(fmaxf(v[0], v[1]), fmaxf(v[2], v[3]));
#pragma unroll
    for (int off = 16; off; off >>= 1) mx = fmaxf(mx, __shfl_xor(mx, off, 32));
    if (lane == 0) redA[wv] = mx;
    __syncthreads();
    mx = redA[0];
#pragma unroll
    for (int w = 1; w < 8; ++w) mx = fmaxf(mx, redA[w]);
#pragma unroll
    for (int i = 0; i < 4; ++i) v[i] /= mx;

    // block sum + sumsq
    float s  = v[0] + v[1] + v[2] + v[3];
    float s2 = v[0] * v[0] + v[1] * v[1] + v[2] * v[2] + v[3] * v[3];
#pragma unroll
    for (int off = 16; off; off >>= 1) {
        s  += __shfl_xor(s,  off, 32);
        s2 += __shfl_xor(s2, off, 32);
    }
    __syncthreads();
    if (lane == 0) { redA[wv] = s; redB[wv] = s2; }
    __syncthreads();
    s = 0.f; s2 = 0.f;
#pragma unroll
    for (int w = 0; w < 8; ++w) { s += redA[w]; s2 += redB[w]; }
    float mean = s / DIMM;
    float var  = s2 / DIMM - mean * mean;
    float inv  = rsqrtf(var + LN_EPS);
#pragma unroll
    for (int i = 0; i < 4; ++i) {
        int c = tid + i * 256;
        store_c(&orow[c], (v[i] - mean) * inv * g[c]);
    }
}

// ---------------------------------------------------------------------------
// 3) bf16 WMMA GEMM:  C[M,N] = A[M,K] @ Bt[N,K]^T      (Bt is B transposed)
//    block = 256 thr (8 waves, 2x4), block tile 128x128, wave tile 64x32.
// ---------------------------------------------------------------------------
template <typename CT>
__global__ __launch_bounds__(256) void gemm_bf16_kernel(const __bf16* __restrict__ A,
                                                        const __bf16* __restrict__ Bt,
                                                        CT* __restrict__ C,
                                                        int M, int N, int K) {
    const int tid  = threadIdx.x;
    const int lane = tid & 31;
    const int wave = tid >> 5;
    const int wm   = wave >> 2;            // 0..1
    const int wn   = wave & 3;             // 0..3
    const int row0 = blockIdx.y * 128 + wm * 64;
    const int col0 = blockIdx.x * 128 + wn * 32;
    const int nlo  = lane & 15;

    f32x8 acc[4][2];
#pragma unroll
    for (int mt = 0; mt < 4; ++mt)
#pragma unroll
        for (int nt = 0; nt < 2; ++nt)
#pragma unroll
            for (int r = 0; r < 8; ++r) acc[mt][nt][r] = 0.f;

    for (int k = 0; k < K; k += 32) {
        // prefetch next K-step (global_prefetch_b8); streaming A benefits most
        if (k + 32 < K) {
            __builtin_prefetch(A + (size_t)(row0 + (lane & 3) * 16) * K + k + 32, 0, 1);
            __builtin_prefetch(Bt + (size_t)(col0 + (lane & 1) * 16) * K + k + 32, 0, 1);
        }
        bf16x16 b0 = load_frag(Bt + (size_t)(col0 + 0)  * K + k, K, lane);
        bf16x16 b1 = load_frag(Bt + (size_t)(col0 + 16) * K + k, K, lane);
#pragma unroll
        for (int mt = 0; mt < 4; ++mt) {
            bf16x16 a = load_frag(A + (size_t)(row0 + mt * 16) * K + k, K, lane);
            acc[mt][0] = wmma_bf16(a, b0, acc[mt][0]);
            acc[mt][1] = wmma_bf16(a, b1, acc[mt][1]);
        }
    }

#pragma unroll
    for (int mt = 0; mt < 4; ++mt)
#pragma unroll
        for (int nt = 0; nt < 2; ++nt)
#pragma unroll
            for (int r = 0; r < 8; ++r) {
                int rr = row0 + mt * 16 + ((lane < 16) ? r : r + 8);
                int cc = col0 + nt * 16 + nlo;
                store_c(&C[(size_t)rr * N + cc], acc[mt][nt][r]);
            }
}

// ---------------------------------------------------------------------------
// 4) repack kv GEMM output into head-major K and transposed V, insert null
//    token at key 0, zero the padding tail.
//    Kh: [b,h,KV_PAD,64]   Vt: [b,h,64,KV_PAD]
// ---------------------------------------------------------------------------
__global__ void repack_kv_kernel(const __bf16* __restrict__ kv,
                                 const float* __restrict__ null_kv,
                                 __bf16* __restrict__ Kh, __bf16* __restrict__ Vt) {
    size_t i = (size_t)blockIdx.x * 256 + threadIdx.x;   // b*h*KV_PAD*64 exact
    int d = (int)(i & 63);
    size_t t = i >> 6;
    int key = (int)(t % KV_PAD); t /= KV_PAD;
    int h = (int)(t % HEADS);
    int b = (int)(t / HEADS);
    float kval, vval;
    if (key == 0) {
        kval = null_kv[d];
        vval = null_kv[DHEAD + d];
    } else if (key <= SEQ) {
        size_t rowoff = ((size_t)b * SEQ + (key - 1)) * (2 * DIMM);
        kval = (float)kv[rowoff + h * DHEAD + d];
        vval = (float)kv[rowoff + DIMM + h * DHEAD + d];
    } else {
        kval = 0.f; vval = 0.f;
    }
    size_t bh = (size_t)b * HEADS + h;
    Kh[(bh * KV_PAD + key) * DHEAD + d] = (__bf16)kval;
    Vt[(bh * DHEAD + d) * KV_PAD + key] = (__bf16)vval;
}

// ---------------------------------------------------------------------------
// 5) flash attention: each wave owns 16 query rows of one (b,h); streams keys
//    32 at a time. S via 2 WMMA (K-dim 64). Row max via DPP reduction.
//    Row sums of P via an extra WMMA against an all-ones B fragment (P @ 1).
//    P bounced through LDS (accumulator layout -> A layout), then 4 WMMA P·V.
//    grid = (SEQ/128, HEADS, BATCH), block = 256 (8 waves)
// ---------------------------------------------------------------------------
__global__ __launch_bounds__(256) void attn_kernel(const __bf16* __restrict__ Q,
                                                   const __bf16* __restrict__ Kh,
                                                   const __bf16* __restrict__ Vt,
                                                   __bf16* __restrict__ O) {
    const int tid  = threadIdx.x;
    const int lane = tid & 31;
    const int wave = tid >> 5;
    const int nlo  = lane & 15;
    const int b = blockIdx.z, h = blockIdx.y;
    const int q0 = blockIdx.x * 128 + wave * 16;

    const size_t qbase = ((size_t)b * SEQ + q0) * DIMM + h * DHEAD;
    bf16x16 qa0 = load_frag(Q + qbase,      DIMM, lane);  // d 0..31
    bf16x16 qa1 = load_frag(Q + qbase + 32, DIMM, lane);  // d 32..63
    const __bf16* Kb = Kh + (size_t)(b * HEADS + h) * KV_PAD * DHEAD;
    const __bf16* Vb = Vt + (size_t)(b * HEADS + h) * DHEAD * KV_PAD;

    // all-ones B fragment: row sums of P come from P @ ones as one WMMA
    bf16x16 ones;
#pragma unroll
    for (int i = 0; i < 16; ++i) ones[i] = (__bf16)1.0f;

    f32x8 acc[4];
    f32x8 accl;            // row sums (replicated across the 16-lane group)
    float mrow[8];
#pragma unroll
    for (int r = 0; r < 8; ++r) { mrow[r] = -1e30f; accl[r] = 0.f; }
#pragma unroll
    for (int j = 0; j < 4; ++j)
#pragma unroll
        for (int r = 0; r < 8; ++r) acc[j][r] = 0.f;

    __shared__ __align__(16) __bf16 Pt[8][16][32];

    for (int kb = 0; kb < KV_PAD; kb += 32) {
        // ---- S = q @ k^T for 32 keys (two 16-key chunks) ----
        f32x8 s[2];
#pragma unroll
        for (int c = 0; c < 2; ++c) {
            const __bf16* kp = Kb + (size_t)(kb + c * 16) * DHEAD;
            bf16x16 k0 = load_frag(kp,      DHEAD, lane);
            bf16x16 k1 = load_frag(kp + 32, DHEAD, lane);
            f32x8 z;
#pragma unroll
            for (int r = 0; r < 8; ++r) z[r] = 0.f;
            z    = wmma_bf16(qa0, k0, z);
            s[c] = wmma_bf16(qa1, k1, z);
        }
        // ---- mask padded keys ----
#pragma unroll
        for (int c = 0; c < 2; ++c)
            if (kb + c * 16 + nlo >= KV_LEN) {
#pragma unroll
                for (int r = 0; r < 8; ++r) s[c][r] = -1e30f;
            }
        // ---- online softmax: DPP row max, exp, rescale ----
#pragma unroll
        for (int r = 0; r < 8; ++r) {
            float mx   = row16_max(fmaxf(s[0][r], s[1][r]));
            float mnew = fmaxf(mrow[r], mx);
            float fold = __expf(mrow[r] - mnew);
            mrow[r] = mnew;
            float p0 = __expf(s[0][r] - mnew);
            float p1 = __expf(s[1][r] - mnew);
            accl[r] *= fold;
#pragma unroll
            for (int j = 0; j < 4; ++j) acc[j][r] *= fold;
            int prow = (lane < 16) ? r : r + 8;
            Pt[wave][prow][nlo]      = (__bf16)p0;
            Pt[wave][prow][16 + nlo] = (__bf16)p1;
        }
        // wave-private LDS bounce: D-layout -> A-layout
        asm volatile("s_wait_dscnt 0" ::: "memory");
        bf16x16 pa = load_frag(&Pt[wave][0][0], 32, lane);
        // ---- l += P @ ones ; out += P @ V ----
        accl = wmma_bf16(pa, ones, accl);
#pragma unroll
        for (int j = 0; j < 4; ++j) {
            bf16x16 vb = load_frag(Vb + (size_t)(j * 16) * KV_PAD + kb, KV_PAD, lane);
            acc[j] = wmma_bf16(pa, vb, acc[j]);
        }
    }

    // ---- epilogue: normalize and write [b, n, h*64 + d] ----
#pragma unroll
    for (int r = 0; r < 8; ++r) {
        float inv = 1.f / accl[r];
        int rr = q0 + ((lane < 16) ? r : r + 8);
        size_t obase = ((size_t)b * SEQ + rr) * DIMM + h * DHEAD + nlo;
#pragma unroll
        for (int j = 0; j < 4; ++j)
            O[obase + j * 16] = (__bf16)(acc[j][r] * inv);
    }
}

// ---------------------------------------------------------------------------
// launch
// ---------------------------------------------------------------------------
extern "C" void kernel_launch(void* const* d_in, const int* in_sizes, int n_in,
                              void* d_out, int out_size, void* d_ws, size_t ws_size,
                              hipStream_t stream) {
    (void)in_sizes; (void)n_in; (void)out_size; (void)ws_size;
    const float* x       = (const float*)d_in[0];
    const float* ctx     = (const float*)d_in[1];
    const float* Wq      = (const float*)d_in[2];
    const float* Wkv     = (const float*)d_in[3];
    const float* Wo      = (const float*)d_in[4];
    const float* g_in    = (const float*)d_in[5];
    const float* g_out   = (const float*)d_in[6];
    const float* null_kv = (const float*)d_in[7];
    float* out = (float*)d_out;

    char* ws = (char*)d_ws;
    size_t off = 0;
    auto alloc = [&](size_t bytes) -> char* {
        char* p = ws + off;
        off += (bytes + 255) & ~(size_t)255;
        return p;
    };
    __bf16* xn    = (__bf16*)alloc((size_t)ROWS * DIMM * 2);       // also attn_out
    __bf16* ctxb  = (__bf16*)alloc((size_t)ROWS * DIMM * 2);
    __bf16* Wq_t  = (__bf16*)alloc((size_t)DIMM * DIMM * 2);
    __bf16* Wkv_t = (__bf16*)alloc((size_t)2 * DIMM * DIMM * 2);
    __bf16* Wo_t  = (__bf16*)alloc((size_t)DIMM * DIMM * 2);
    __bf16* qbuf  = (__bf16*)alloc((size_t)ROWS * DIMM * 2);
    __bf16* kvbuf = (__bf16*)alloc((size_t)ROWS * 2 * DIMM * 2);   // also out_f32
    __bf16* KhB   = (__bf16*)alloc((size_t)BATCH * HEADS * KV_PAD * DHEAD * 2);
    __bf16* VtB   = (__bf16*)alloc((size_t)BATCH * HEADS * DHEAD * KV_PAD * 2);
    __bf16* attn_out = xn;             // xn dead after q-GEMM
    float*  out_f32  = (float*)kvbuf;  // kvbuf dead after repack; sizes match

    // 1) weights -> bf16 transposed (SCALE folded into Wq)
    wcvt_t_kernel<<<(DIMM * DIMM + 255) / 256, 256, 0, stream>>>(Wq, Wq_t, DIMM, DIMM, SCALE_Q);
    wcvt_t_kernel<<<(2 * DIMM * DIMM + 255) / 256, 256, 0, stream>>>(Wkv, Wkv_t, DIMM, 2 * DIMM, 1.0f);
    wcvt_t_kernel<<<(DIMM * DIMM + 255) / 256, 256, 0, stream>>>(Wo, Wo_t, DIMM, DIMM, 1.0f);

    // 2) stable LN on x -> bf16 ; context -> bf16
    ln_stable_kernel<__bf16><<<ROWS, 256, 0, stream>>>(x, g_in, xn);
    cvt_bf16_kernel<<<(ROWS * DIMM + 255) / 256, 256, 0, stream>>>(ctx, ctxb, (size_t)ROWS * DIMM);

    // 3) projections
    gemm_bf16_kernel<__bf16><<<dim3(DIMM / 128, ROWS / 128), 256, 0, stream>>>(
        xn, Wq_t, qbuf, ROWS, DIMM, DIMM);
    gemm_bf16_kernel<__bf16><<<dim3(2 * DIMM / 128, ROWS / 128), 256, 0, stream>>>(
        ctxb, Wkv_t, kvbuf, ROWS, 2 * DIMM, DIMM);

    // 4) repack k/v (+null token, +zero pad)
    {
        size_t tot = (size_t)BATCH * HEADS * KV_PAD * DHEAD;
        repack_kv_kernel<<<(unsigned)(tot / 256), 256, 0, stream>>>(kvbuf, null_kv, KhB, VtB);
    }

    // 5) attention
    attn_kernel<<<dim3(SEQ / 128, HEADS, BATCH), 256, 0, stream>>>(qbuf, KhB, VtB, attn_out);

    // 6) output projection (f32) + final stable LN
    gemm_bf16_kernel<float><<<dim3(DIMM / 128, ROWS / 128), 256, 0, stream>>>(
        attn_out, Wo_t, out_f32, ROWS, DIMM, DIMM);
    ln_stable_kernel<float><<<ROWS, 256, 0, stream>>>(out_f32, g_out, out);
}
